// MambaMoELayer_50921132261644
// MI455X (gfx1250) — matrix operand order repaired and checked by reading, not compiled, not run on hardware
//
#include <hip/hip_runtime.h>
#include <hip/hip_bf16.h>

// ---------------------------------------------------------------------------
// MI455X (gfx1250) implementation.
// All big GEMMs use V_WMMA_F32_16X16X32_BF16 (bf16 in, f32 accumulate):
// roofline says the layer is HBM-bound (~226MB expert weights @ 23.3 TB/s)
// only if GEMM compute uses the K=32 bf16 WMMA path; f32 16x16x4 WMMA would
// make it compute-bound. Weights stay f32 in HBM, converted per-tile in LDS.
// MoE tokens are grouped by expert so each expert weight panel streams once.
// ---------------------------------------------------------------------------

typedef __bf16 bf16_t;
typedef __bf16 v16bf __attribute__((ext_vector_type(16)));
typedef __bf16 v8bf  __attribute__((ext_vector_type(8)));
typedef float  v8f   __attribute__((ext_vector_type(8)));

#define D_MODEL  768
#define D_STATE  16
#define D_INNER  1536
#define D_FF     3072
#define N_EXP    8
#define T_LEN    1024
#define NTOK     2048      // B_SZ * T_LEN
#define EPSF     1e-6f

// Tile config: block = 256 threads = 8 wave32 (4x2), block tile 128x64,
// wave tile 32x32 (2x2 WMMA 16x16 tiles), K-step 32.
#define BM 128
#define BN 64
#define BK 32
#define LDSA_S 40   // padded stride (elems) for A tile rows (80B, 16B-aligned)
#define LDSB_S 40   // padded stride (elems) for B^T tile rows

union FragU { v16bf v; v8bf h[2]; };

__device__ inline v8f vzero8() {
  v8f z;
  #pragma unroll
  for (int j = 0; j < 8; ++j) z[j] = 0.0f;
  return z;
}

__device__ inline v8f wmma_bf16(v16bf a, v16bf b, v8f c) {
  return __builtin_amdgcn_wmma_f32_16x16x32_bf16(false, a, false, b,
                                                 (short)0, c, false, false);
}

// ---- LDS staging -----------------------------------------------------------

// A tile (BMxBK) from bf16 row-major global, rows beyond mvalid -> zero.
__device__ inline void stage_A_bf16(bf16_t* __restrict__ lA,
                                    const bf16_t* __restrict__ A, int lda,
                                    int mvalid, int k0, int tid) {
  #pragma unroll
  for (int i = 0; i < 2; ++i) {
    int idx = i * 256 + tid;
    int row = idx >> 2;
    int cb  = idx & 3;
    uint4 v = make_uint4(0u, 0u, 0u, 0u);
    if (row < mvalid)
      v = *(const uint4*)(A + (size_t)row * lda + k0 + cb * 8);
    *(uint4*)(lA + row * LDSA_S + cb * 8) = v;
  }
}

// A tile from f32 global with row gather + per-column scale (g_exp), -> bf16.
__device__ inline void stage_A_f32_gather(bf16_t* __restrict__ lA,
                                          const float* __restrict__ X, int lda,
                                          const int* __restrict__ rows,
                                          int mvalid, int k0,
                                          const float* __restrict__ scale,
                                          int tid) {
  #pragma unroll
  for (int i = 0; i < 4; ++i) {
    int idx = i * 256 + tid;
    int row = idx >> 3;
    int col = (idx & 7) * 4;
    bf16_t o0 = (bf16_t)0.f, o1 = (bf16_t)0.f, o2 = (bf16_t)0.f, o3 = (bf16_t)0.f;
    if (row < mvalid) {
      const float4 v = *(const float4*)(X + (size_t)rows[row] * lda + k0 + col);
      o0 = (bf16_t)(v.x * scale[k0 + col + 0]);
      o1 = (bf16_t)(v.y * scale[k0 + col + 1]);
      o2 = (bf16_t)(v.z * scale[k0 + col + 2]);
      o3 = (bf16_t)(v.w * scale[k0 + col + 3]);
    }
    bf16_t* d = lA + row * LDSA_S + col;
    d[0] = o0; d[1] = o1; d[2] = o2; d[3] = o3;
  }
}

// B tile (BKxBN) from f32 row-major global K x N, stored transposed (N-major,
// K contiguous) as bf16 so fragment loads are two contiguous 16B ds reads.
__device__ inline void stage_B_f32(bf16_t* __restrict__ lB,
                                   const float* __restrict__ B, int ldb,
                                   int k0, int n0, int tid) {
  #pragma unroll
  for (int i = 0; i < 2; ++i) {
    int idx = i * 256 + tid;
    int kr  = idx >> 4;
    int col = (idx & 15) * 4;
    const float* src = B + (size_t)(k0 + kr) * ldb + n0 + col;
    const float4 v = *(const float4*)src;
    // speculative prefetch of next K-step tile (emits global_prefetch_b8)
    __builtin_prefetch(src + (size_t)BK * ldb, 0, 1);
    lB[(col + 0) * LDSB_S + kr] = (bf16_t)v.x;
    lB[(col + 1) * LDSB_S + kr] = (bf16_t)v.y;
    lB[(col + 2) * LDSB_S + kr] = (bf16_t)v.z;
    lB[(col + 3) * LDSB_S + kr] = (bf16_t)v.w;
  }
}

// ---- WMMA fragment loads (documented CDNA5 16-bit layouts) -----------------
// A 16x32: lanes 0-15 = M, K {0..7,16..23}; lanes 16-31 = same M, K {8..15,24..31}
__device__ inline v16bf load_fragA(const bf16_t* __restrict__ lA, int rbase, int lane) {
  int r  = rbase + (lane & 15);
  int kh = ((lane >> 4) & 1) << 3;     // 0 or 8
  FragU f;
  f.h[0] = *(const v8bf*)&lA[r * LDSA_S + kh];
  f.h[1] = *(const v8bf*)&lA[r * LDSA_S + 16 + kh];
  return f.v;
}
// B 32x16: lanes 0-15 = N, K {0..7,16..23}; lanes 16-31 = same N, K {8..15,24..31}
__device__ inline v16bf load_fragB(const bf16_t* __restrict__ lB, int cbase, int lane) {
  int c  = cbase + (lane & 15);
  int kh = ((lane >> 4) & 1) << 3;     // 0 or 8
  FragU f;
  f.h[0] = *(const v8bf*)&lB[c * LDSB_S + kh];
  f.h[1] = *(const v8bf*)&lB[c * LDSB_S + 16 + kh];
  return f.v;
}

// ===========================================================================
// Kernel 1: RMSNorm(x, g) -> bf16 activations
// ===========================================================================
__global__ __launch_bounds__(256) void k_rms_bf16(const float* __restrict__ x,
                                                  const float* __restrict__ g,
                                                  bf16_t* __restrict__ hbf) {
  const int n = blockIdx.x, tid = threadIdx.x;
  __shared__ float red[256];
  const float* xr = x + (size_t)n * D_MODEL;
  float v[3]; float ss = 0.f;
  #pragma unroll
  for (int i = 0; i < 3; ++i) { v[i] = xr[tid + i * 256]; ss += v[i] * v[i]; }
  red[tid] = ss; __syncthreads();
  for (int s = 128; s > 0; s >>= 1) { if (tid < s) red[tid] += red[tid + s]; __syncthreads(); }
  const float r = rsqrtf(red[0] / (float)D_MODEL + EPSF);
  #pragma unroll
  for (int i = 0; i < 3; ++i) {
    int k = tid + i * 256;
    hbf[(size_t)n * D_MODEL + k] = (bf16_t)(v[i] * r * g[k]);
  }
}

// ===========================================================================
// Kernel 2: xz = h_bf16 (2048x768) @ w_in (768x3072)  [f32 out]
// ===========================================================================
__global__ __launch_bounds__(256) void k_gemm_xin(const bf16_t* __restrict__ A,
                                                  const float* __restrict__ B,
                                                  float* __restrict__ C) {
  __shared__ bf16_t lA[BM * LDSA_S];
  __shared__ bf16_t lB[BN * LDSB_S];
  const int tid = threadIdx.x, lane = tid & 31, wid = tid >> 5;
  const int wm = (wid & 3) * 32, wn = (wid >> 2) * 32;
  const int m0 = blockIdx.x * BM, n0 = blockIdx.y * BN;
  v8f acc[2][2];
  acc[0][0] = vzero8(); acc[0][1] = vzero8(); acc[1][0] = vzero8(); acc[1][1] = vzero8();
  for (int k0 = 0; k0 < D_MODEL; k0 += BK) {
    stage_A_bf16(lA, A + (size_t)m0 * D_MODEL, D_MODEL, BM, k0, tid);
    stage_B_f32(lB, B, 2 * D_INNER, k0, n0, tid);
    __syncthreads();
    v16bf a0 = load_fragA(lA, wm, lane),      a1 = load_fragA(lA, wm + 16, lane);
    v16bf b0 = load_fragB(lB, wn, lane),      b1 = load_fragB(lB, wn + 16, lane);
    acc[0][0] = wmma_bf16(a0, b0, acc[0][0]);
    acc[0][1] = wmma_bf16(a0, b1, acc[0][1]);
    acc[1][0] = wmma_bf16(a1, b0, acc[1][0]);
    acc[1][1] = wmma_bf16(a1, b1, acc[1][1]);
    __syncthreads();
  }
  #pragma unroll
  for (int mt = 0; mt < 2; ++mt)
    #pragma unroll
    for (int nt = 0; nt < 2; ++nt) {
      const int col = n0 + wn + nt * 16 + (lane & 15);
      const int rb  = m0 + wm + mt * 16 + ((lane >> 4) << 3);
      #pragma unroll
      for (int j = 0; j < 8; ++j)
        C[(size_t)(rb + j) * (2 * D_INNER) + col] = acc[mt][nt][j];
    }
}

// ===========================================================================
// Kernel 3: depthwise causal conv (D_CONV=4) + SiLU ; x_s = xz[:, :1536]
// ===========================================================================
__global__ __launch_bounds__(256) void k_conv(const float* __restrict__ xz,
                                              const float* __restrict__ cw,
                                              const float* __restrict__ cb,
                                              float* __restrict__ xc) {
  const int idx = blockIdx.x * 256 + threadIdx.x;
  if (idx >= NTOK * D_INNER) return;
  const int d = idx % D_INNER;
  const int n = idx / D_INNER;
  const int t = n % T_LEN, b = n / T_LEN;
  float acc = cb[d];
  #pragma unroll
  for (int k = 0; k < 4; ++k) {
    const int tt = t + k - 3;
    if (tt >= 0) acc += xz[(size_t)(b * T_LEN + tt) * (2 * D_INNER) + d] * cw[d * 4 + k];
  }
  xc[idx] = acc / (1.f + __expf(-acc));   // SiLU
}

// ===========================================================================
// Kernel 4: params = xc (2048x1536) @ w_xproj (1536x33)   [tiny GEMM]
// ===========================================================================
__global__ __launch_bounds__(256) void k_xproj(const float* __restrict__ xc,
                                               const float* __restrict__ w,
                                               float* __restrict__ params) {
  const int idx = blockIdx.x * 256 + threadIdx.x;
  if (idx >= NTOK * 33) return;
  const int n = idx / 33, c = idx % 33;
  const float* xr = xc + (size_t)n * D_INNER;
  float acc = 0.f;
  for (int k = 0; k < D_INNER; ++k) acc += xr[k] * w[k * 33 + c];
  params[idx] = acc;
}

// ===========================================================================
// Kernel 5: selective SSM scan over T per (b,d); emits y*silu(z) as bf16
// ===========================================================================
__global__ __launch_bounds__(256) void k_scan(const float* __restrict__ xz,
                                              const float* __restrict__ xc,
                                              const float* __restrict__ params,
                                              const float* __restrict__ w_dt,
                                              const float* __restrict__ b_dt,
                                              const float* __restrict__ A_log,
                                              const float* __restrict__ D_skip,
                                              bf16_t* __restrict__ ybf) {
  const int idx = blockIdx.x * 256 + threadIdx.x;
  if (idx >= 2 * D_INNER) return;
  const int d = idx % D_INNER, b = idx / D_INNER;
  float Av[D_STATE];
  #pragma unroll
  for (int s = 0; s < D_STATE; ++s) Av[s] = -__expf(A_log[d * D_STATE + s]);
  const float wdt = w_dt[d], bdt = b_dt[d], dsk = D_skip[d];
  float h[D_STATE];
  #pragma unroll
  for (int s = 0; s < D_STATE; ++s) h[s] = 0.f;
  for (int t = 0; t < T_LEN; ++t) {
    const int n = b * T_LEN + t;
    const float* pr = params + (size_t)n * 33;
    float dt = pr[32] * wdt + bdt;
    dt = (dt > 20.f) ? dt : log1pf(__expf(dt));           // softplus
    const float u = xc[(size_t)n * D_INNER + d];
    float y = 0.f;
    #pragma unroll
    for (int s = 0; s < D_STATE; ++s) {
      const float Ab = __expf(dt * Av[s]);
      h[s] = Ab * h[s] + dt * pr[s] * u;
      y += h[s] * pr[D_STATE + s];
    }
    y += u * dsk;
    const float z = xz[(size_t)n * (2 * D_INNER) + D_INNER + d];
    y *= z / (1.f + __expf(-z));                          // * silu(z)
    ybf[(size_t)n * D_INNER + d] = (bf16_t)y;
  }
}

// ===========================================================================
// Kernel 6: x2 = residual + y_bf16 (2048x1536) @ w_out (1536x768); out=x2 too
// ===========================================================================
__global__ __launch_bounds__(256) void k_gemm_out(const bf16_t* __restrict__ A,
                                                  const float* __restrict__ B,
                                                  const float* __restrict__ resid,
                                                  float* __restrict__ x2,
                                                  float* __restrict__ outp) {
  __shared__ bf16_t lA[BM * LDSA_S];
  __shared__ bf16_t lB[BN * LDSB_S];
  const int tid = threadIdx.x, lane = tid & 31, wid = tid >> 5;
  const int wm = (wid & 3) * 32, wn = (wid >> 2) * 32;
  const int m0 = blockIdx.x * BM, n0 = blockIdx.y * BN;
  v8f acc[2][2];
  acc[0][0] = vzero8(); acc[0][1] = vzero8(); acc[1][0] = vzero8(); acc[1][1] = vzero8();
  for (int k0 = 0; k0 < D_INNER; k0 += BK) {
    stage_A_bf16(lA, A + (size_t)m0 * D_INNER, D_INNER, BM, k0, tid);
    stage_B_f32(lB, B, D_MODEL, k0, n0, tid);
    __syncthreads();
    v16bf a0 = load_fragA(lA, wm, lane),      a1 = load_fragA(lA, wm + 16, lane);
    v16bf b0 = load_fragB(lB, wn, lane),      b1 = load_fragB(lB, wn + 16, lane);
    acc[0][0] = wmma_bf16(a0, b0, acc[0][0]);
    acc[0][1] = wmma_bf16(a0, b1, acc[0][1]);
    acc[1][0] = wmma_bf16(a1, b0, acc[1][0]);
    acc[1][1] = wmma_bf16(a1, b1, acc[1][1]);
    __syncthreads();
  }
  #pragma unroll
  for (int mt = 0; mt < 2; ++mt)
    #pragma unroll
    for (int nt = 0; nt < 2; ++nt) {
      const int col = n0 + wn + nt * 16 + (lane & 15);
      const int rb  = m0 + wm + mt * 16 + ((lane >> 4) << 3);
      #pragma unroll
      for (int j = 0; j < 8; ++j) {
        const size_t o = (size_t)(rb + j) * D_MODEL + col;
        const float v = resid[o] + acc[mt][nt][j];
        x2[o] = v;
        outp[o] = v;   // top-2 softmax weights sum to 1 -> out = x2 + sum w*ffn
      }
    }
}

// ===========================================================================
// Kernel 7: router -- rms_x, top-2 expert selection, per-expert token lists
// meta[0..7] = counts, meta[8..15] = exclusive offsets (filled by k_offs)
// ===========================================================================
__global__ __launch_bounds__(256) void k_router(const float* __restrict__ x2,
                                                const float* __restrict__ g_moe,
                                                const float* __restrict__ w_router,
                                                float* __restrict__ rmsx,
                                                int* __restrict__ meta,
                                                int* __restrict__ toklist,
                                                float* __restrict__ wlist) {
  const int n = blockIdx.x, tid = threadIdx.x;
  __shared__ float red[256];
  __shared__ float srow[D_MODEL];
  __shared__ float sc[N_EXP];
  const float* xr = x2 + (size_t)n * D_MODEL;
  float ss = 0.f;
  for (int k = tid; k < D_MODEL; k += 256) { float v = xr[k]; srow[k] = v; ss += v * v; }
  red[tid] = ss; __syncthreads();
  for (int s = 128; s > 0; s >>= 1) { if (tid < s) red[tid] += red[tid + s]; __syncthreads(); }
  const float r = rsqrtf(red[0] / (float)D_MODEL + EPSF);
  for (int k = tid; k < D_MODEL; k += 256) rmsx[(size_t)n * D_MODEL + k] = srow[k] * r;
  if (tid < N_EXP) {
    float acc = 0.f;
    for (int k = 0; k < D_MODEL; ++k)
      acc += srow[k] * r * g_moe[k] * w_router[k * N_EXP + tid];
    sc[tid] = acc;
  }
  __syncthreads();
  if (tid == 0) {
    int i1 = 0; float v1 = sc[0];
    for (int i = 1; i < N_EXP; ++i) if (sc[i] > v1) { v1 = sc[i]; i1 = i; }
    int i2 = -1; float v2 = -1e30f;
    for (int i = 0; i < N_EXP; ++i) if (i != i1 && sc[i] > v2) { v2 = sc[i]; i2 = i; }
    const float wA = 1.f / (1.f + __expf(v2 - v1));   // softmax over {v1,v2}
    const float wB = 1.f - wA;
    int p1 = atomicAdd(&meta[i1], 1);
    toklist[i1 * NTOK + p1] = n; wlist[i1 * NTOK + p1] = wA;
    int p2 = atomicAdd(&meta[i2], 1);
    toklist[i2 * NTOK + p2] = n; wlist[i2 * NTOK + p2] = wB;
  }
}

__global__ void k_zero_meta(int* meta) { if (threadIdx.x < 16) meta[threadIdx.x] = 0; }
__global__ void k_offs(int* meta) {
  if (threadIdx.x == 0) { int s = 0; for (int i = 0; i < N_EXP; ++i) { meta[8 + i] = s; s += meta[i]; } }
}

// ===========================================================================
// Kernel 8 (MoE stage 1): act = silu(he@w1[e]) * (he@w3[e]), he gathered rows
// of rms_x scaled by g_exp[e]; stored bf16 at per-expert contiguous slots.
// ===========================================================================
__global__ __launch_bounds__(256) void k_moe1(const float* __restrict__ rmsx,
                                              const float* __restrict__ g_exp,
                                              const float* __restrict__ w1,
                                              const float* __restrict__ w3,
                                              const int* __restrict__ meta,
                                              const int* __restrict__ toklist,
                                              bf16_t* __restrict__ act) {
  const int e   = blockIdx.z;
  const int cnt = meta[e];
  const int m0  = blockIdx.x * BM;
  if (m0 >= cnt) return;
  const int n0  = blockIdx.y * BN;
  const int off = meta[8 + e];
  const int mv  = min(cnt - m0, BM);
  const int* rows = toklist + e * NTOK + m0;
  const float* ge = g_exp + (size_t)e * D_MODEL;
  const float* B1 = w1 + (size_t)e * D_MODEL * D_FF;
  const float* B3 = w3 + (size_t)e * D_MODEL * D_FF;

  __shared__ bf16_t lA[BM * LDSA_S];
  __shared__ bf16_t lB1[BN * LDSB_S];
  __shared__ bf16_t lB3[BN * LDSB_S];
  const int tid = threadIdx.x, lane = tid & 31, wid = tid >> 5;
  const int wm = (wid & 3) * 32, wn = (wid >> 2) * 32;
  v8f acc1[2][2], acc3[2][2];
  #pragma unroll
  for (int i = 0; i < 2; ++i)
    #pragma unroll
    for (int j = 0; j < 2; ++j) { acc1[i][j] = vzero8(); acc3[i][j] = vzero8(); }

  for (int k0 = 0; k0 < D_MODEL; k0 += BK) {
    stage_A_f32_gather(lA, rmsx, D_MODEL, rows, mv, k0, ge, tid);
    stage_B_f32(lB1, B1, D_FF, k0, n0, tid);
    stage_B_f32(lB3, B3, D_FF, k0, n0, tid);
    __syncthreads();
    v16bf a0  = load_fragA(lA, wm, lane),       a1  = load_fragA(lA, wm + 16, lane);
    v16bf b10 = load_fragB(lB1, wn, lane),      b11 = load_fragB(lB1, wn + 16, lane);
    v16bf b30 = load_fragB(lB3, wn, lane),      b31 = load_fragB(lB3, wn + 16, lane);
    acc1[0][0] = wmma_bf16(a0, b10, acc1[0][0]);
    acc1[0][1] = wmma_bf16(a0, b11, acc1[0][1]);
    acc1[1][0] = wmma_bf16(a1, b10, acc1[1][0]);
    acc1[1][1] = wmma_bf16(a1, b11, acc1[1][1]);
    acc3[0][0] = wmma_bf16(a0, b30, acc3[0][0]);
    acc3[0][1] = wmma_bf16(a0, b31, acc3[0][1]);
    acc3[1][0] = wmma_bf16(a1, b30, acc3[1][0]);
    acc3[1][1] = wmma_bf16(a1, b31, acc3[1][1]);
    __syncthreads();
  }
  #pragma unroll
  for (int mt = 0; mt < 2; ++mt)
    #pragma unroll
    for (int nt = 0; nt < 2; ++nt) {
      const int col = n0 + wn + nt * 16 + (lane & 15);
      const int rb  = wm + mt * 16 + ((lane >> 4) << 3);
      #pragma unroll
      for (int j = 0; j < 8; ++j) {
        const int row = rb + j;
        if (row < mv) {
          const float a = acc1[mt][nt][j];
          const float b = acc3[mt][nt][j];
          const float v = (a / (1.f + __expf(-a))) * b;   // silu(a)*b
          act[(size_t)(off + m0 + row) * D_FF + col] = (bf16_t)v;
        }
      }
    }
}

// ===========================================================================
// Kernel 9 (MoE stage 2): out[tok] += w_slot * (act_slot @ w2[e])
// ===========================================================================
__global__ __launch_bounds__(256) void k_moe2(const bf16_t* __restrict__ act,
                                              const float* __restrict__ w2,
                                              const int* __restrict__ meta,
                                              const int* __restrict__ toklist,
                                              const float* __restrict__ wlist,
                                              float* __restrict__ outp) {
  const int e   = blockIdx.z;
  const int cnt = meta[e];
  const int m0  = blockIdx.x * BM;
  if (m0 >= cnt) return;
  const int n0  = blockIdx.y * BN;
  const int off = meta[8 + e];
  const int mv  = min(cnt - m0, BM);
  const bf16_t* Abase = act + (size_t)(off + m0) * D_FF;
  const float* B = w2 + (size_t)e * D_FF * D_MODEL;

  __shared__ bf16_t lA[BM * LDSA_S];
  __shared__ bf16_t lB[BN * LDSB_S];
  const int tid = threadIdx.x, lane = tid & 31, wid = tid >> 5;
  const int wm = (wid & 3) * 32, wn = (wid >> 2) * 32;
  v8f acc[2][2];
  acc[0][0] = vzero8(); acc[0][1] = vzero8(); acc[1][0] = vzero8(); acc[1][1] = vzero8();

  for (int k0 = 0; k0 < D_FF; k0 += BK) {
    stage_A_bf16(lA, Abase, D_FF, mv, k0, tid);
    stage_B_f32(lB, B, D_MODEL, k0, n0, tid);
    __syncthreads();
    v16bf a0 = load_fragA(lA, wm, lane),      a1 = load_fragA(lA, wm + 16, lane);
    v16bf b0 = load_fragB(lB, wn, lane),      b1 = load_fragB(lB, wn + 16, lane);
    acc[0][0] = wmma_bf16(a0, b0, acc[0][0]);
    acc[0][1] = wmma_bf16(a0, b1, acc[0][1]);
    acc[1][0] = wmma_bf16(a1, b0, acc[1][0]);
    acc[1][1] = wmma_bf16(a1, b1, acc[1][1]);
    __syncthreads();
  }
  #pragma unroll
  for (int mt = 0; mt < 2; ++mt)
    #pragma unroll
    for (int nt = 0; nt < 2; ++nt) {
      const int col = n0 + wn + nt * 16 + (lane & 15);
      const int rb  = wm + mt * 16 + ((lane >> 4) << 3);
      #pragma unroll
      for (int j = 0; j < 8; ++j) {
        const int row = rb + j;
        if (row < mv) {
          const int   slot = e * NTOK + m0 + row;
          const int   tok  = toklist[slot];
          const float wgt  = wlist[slot];
          atomicAdd(&outp[(size_t)tok * D_MODEL + col], wgt * acc[mt][nt][j]);
        }
      }
    }
}

// ===========================================================================
// Host launcher
// ===========================================================================
extern "C" void kernel_launch(void* const* d_in, const int* in_sizes, int n_in,
                              void* d_out, int out_size, void* d_ws, size_t ws_size,
                              hipStream_t stream) {
  const float* x        = (const float*)d_in[0];
  const float* g_mamba  = (const float*)d_in[1];
  const float* w_in     = (const float*)d_in[2];
  const float* conv_w   = (const float*)d_in[3];
  const float* conv_b   = (const float*)d_in[4];
  const float* w_xproj  = (const float*)d_in[5];
  const float* w_dt     = (const float*)d_in[6];
  const float* b_dt     = (const float*)d_in[7];
  const float* A_log    = (const float*)d_in[8];
  const float* D_skip   = (const float*)d_in[9];
  const float* w_out    = (const float*)d_in[10];
  const float* g_moe    = (const float*)d_in[11];
  const float* w_router = (const float*)d_in[12];
  const float* g_exp    = (const float*)d_in[13];
  const float* w1       = (const float*)d_in[14];
  const float* w3       = (const float*)d_in[15];
  const float* w2       = (const float*)d_in[16];
  float* outp = (float*)d_out;

  // Workspace carve-up (all offsets 16B-aligned).  act aliases xz: xz (z-gate)
  // is dead after k_scan, before k_moe1 writes act.
  char* p = (char*)d_ws;
  bf16_t* h_bf  = (bf16_t*)p;  p += (size_t)NTOK * D_MODEL * 2;        // 3.0 MB
  float*  xz    = (float*)p;   p += (size_t)NTOK * 2 * D_INNER * 4;    // 25.2 MB
  bf16_t* act   = (bf16_t*)xz;                                         // alias (25.2 MB, 2*NTOK rows)
  float*  xc    = (float*)p;   p += (size_t)NTOK * D_INNER * 4;        // 12.6 MB
  float*  params= (float*)p;   p += (size_t)NTOK * 33 * 4;             // 0.26 MB
  bf16_t* ybf   = (bf16_t*)p;  p += (size_t)NTOK * D_INNER * 2;        // 6.3 MB
  float*  x2    = (float*)p;   p += (size_t)NTOK * D_MODEL * 4;        // 6.3 MB
  float*  rmsx  = (float*)p;   p += (size_t)NTOK * D_MODEL * 4;        // 6.3 MB
  int*    meta  = (int*)p;     p += 64;
  int*    toklist = (int*)p;   p += (size_t)N_EXP * NTOK * 4;          // 64 KB
  float*  wlist = (float*)p;   p += (size_t)N_EXP * NTOK * 4;          // 64 KB
  (void)ws_size; (void)in_sizes; (void)n_in; (void)out_size;

  // ---- Mamba block ----
  k_zero_meta<<<1, 32, 0, stream>>>(meta);
  k_rms_bf16<<<NTOK, 256, 0, stream>>>(x, g_mamba, h_bf);
  k_gemm_xin<<<dim3(NTOK / BM, (2 * D_INNER) / BN), 256, 0, stream>>>(h_bf, w_in, xz);
  k_conv<<<(NTOK * D_INNER) / 256, 256, 0, stream>>>(xz, conv_w, conv_b, xc);
  k_xproj<<<(NTOK * 33 + 255) / 256, 256, 0, stream>>>(xc, w_xproj, params);
  k_scan<<<(2 * D_INNER + 255) / 256, 256, 0, stream>>>(xz, xc, params, w_dt, b_dt,
                                                        A_log, D_skip, ybf);
  k_gemm_out<<<dim3(NTOK / BM, D_MODEL / BN), 256, 0, stream>>>(ybf, w_out, x, x2, outp);

  // ---- MoE block ----
  k_router<<<NTOK, 256, 0, stream>>>(x2, g_moe, w_router, rmsx, meta, toklist, wlist);
  k_offs<<<1, 32, 0, stream>>>(meta);
  k_moe1<<<dim3(NTOK / BM, D_FF / BN, N_EXP), 256, 0, stream>>>(rmsx, g_exp, w1, w3,
                                                                meta, toklist, act);
  k_moe2<<<dim3(NTOK / BM, D_MODEL / BN, N_EXP), 256, 0, stream>>>(act, w2, meta,
                                                                   toklist, wlist, outp);
}
